// GridSampleRefiner_76338748719738
// MI455X (gfx1250) — matrix operand order repaired and checked by reading, not compiled, not run on hardware
//
#include <hip/hip_runtime.h>

typedef _Float16 half16 __attribute__((ext_vector_type(16)));
typedef float    float8 __attribute__((ext_vector_type(8)));

#define CC 64
#define HWPROD 65536   // 256*256

// LDS layout constants for kernel 1 (dynamic LDS, carved manually).
// XS_STRIDE = 80 halfs = 160 B per pixel slot -> every 16-channel fragment
// is 32B-aligned, so a half16 deref lowers to two ds_load_b128 directly into
// a contiguous even-aligned VGPR tuple (no v_mov shuffling into WMMA operands).
#define XS_STRIDE 80
#define XS_HALFS  (3 * 258 * XS_STRIDE)  // input slab: 3 rows x (256+2 halo) px
#define LW_HALFS  (2 * 18 * 32 * 16)     // permuted A-fragments
#define XS(row, px, c) (((row) * 258 + (px)) * XS_STRIDE + (c))
#define LW(mt, ks, l, j) ((((mt) * 18 + (ks)) * 32 + (l)) * 16 + (j))
#define K1_LDS_BYTES ((XS_HALFS + LW_HALFS) * 2 + (288 + 32 + 32) * 4)  // ~162 KB < 320 KB WGP LDS

static __device__ __forceinline__ float8 wmma_f16(half16 a, half16 b, float8 c) {
  // v_wmma_f32_16x16x32_f16 : D = A(16x32 f16) x B(32x16 f16) + C(16x16 f32)
  return __builtin_amdgcn_wmma_f32_16x16x32_f16(false, a, false, b, (short)0, c, false, false);
}

static __device__ __forceinline__ float silu(float z) {
  return z / (1.0f + __expf(-z));
}

// -------------------------------------------------------------------------
// Kernel 1: conv3x3(65->32) + BN1 + SiLU + 1x1(32->2) + grid construction.
// Implicit GEMM: M = out channels (2x16 tiles), N = pixels (wave: 32 px of
// one image row = 2 N-tiles), K = 9 taps x 2 c-blocks of 32 via WMMA.
// Input rows staged ONCE in LDS as f16, channel-contiguous, halo pre-zeroed:
// each B fragment is two ds_load_b128 (no global gathers, no masks, no cvt).
// Heatmap channel (c=64) is a VALU rank-1 update into the C fragments.
// Tap loop kept rolled (unroll 1) to stay under 256 VGPRs (no s_set_vgpr_msb).
// -------------------------------------------------------------------------
__global__ __launch_bounds__(256) void k1_offset_grid(
    const float* __restrict__ x, const float* __restrict__ hm,
    const float* __restrict__ w1,
    const float* __restrict__ g1, const float* __restrict__ b1,
    const float* __restrict__ m1, const float* __restrict__ v1,
    const float* __restrict__ w2, const float* __restrict__ bo2,
    float* __restrict__ grid)
{
  extern __shared__ __align__(32) char smem[];
  _Float16* xs   = (_Float16*)smem;             // [3][258][XS_STRIDE] f16 input slab
  _Float16* lw   = xs + XS_HALFS;               // [2][18][32][16] A-fragments
  float*    lw64 = (float*)(lw + LW_HALFS);     // [32][9] heatmap-channel weights
  float*    lsc  = lw64 + 288;                  // [32] folded BN1 scale
  float*    lsh  = lsc + 32;                    // [32] folded BN1 shift
  _Float16* h1s  = xs;                          // aliased after barrier: [8][32][32]

  const int tid  = threadIdx.x;
  const int lane = tid & 31;
  const int wave = tid >> 5;
  const int bb   = blockIdx.x >> 8;   // batch
  const int hh   = blockIdx.x & 255;  // row

  const float* __restrict__ xb = x  + (size_t)bb * CC * HWPROD;
  const float* __restrict__ hb = hm + (size_t)bb * HWPROD;

  // --- stage conv weights into LDS in exact A-fragment layout ---
  for (int e = tid; e < 2 * 18 * 32; e += 256) {
    const int l  = e & 31;
    const int ks = (e >> 5) % 18;
    const int mt = e / (18 * 32);
    const int tap = ks >> 1, cb = ks & 1;
    const int dy = tap / 3, dx = tap - dy * 3;
    const int m  = mt * 16 + (l & 15);
    const int kh = (l < 16) ? 0 : 16;
#pragma unroll
    for (int j = 0; j < 16; ++j) {
      const int k = j + kh; // contiguous K order (k = 16h + j), matches B
      const int c = cb * 32 + k;
      lw[LW(mt, ks, l, j)] = (_Float16)w1[((m * 65 + c) * 3 + dy) * 3 + dx];
    }
  }
  for (int e = tid; e < 32 * 9; e += 256) {
    const int m = e / 9, tap = e - m * 9;
    lw64[m * 9 + tap] = w1[((m * 65 + 64) * 3 + tap / 3) * 3 + (tap % 3)];
  }
  if (tid < 32) {
    const float inv = g1[tid] * rsqrtf(v1[tid] + 1e-5f);
    lsc[tid] = inv;
    lsh[tid] = b1[tid] - m1[tid] * inv;
  }

  // --- stage 3 input rows (64 ch x 256 px) as f16, channel-contiguous ---
  // float4 global loads: fully coalesced, 4x fewer vmem ops.
  for (int e = tid; e < 3 * 64 * 64; e += 256) {
    const int row = e >> 12;          // / (64*64)
    const int rem = e & 4095;
    const int c   = rem >> 6;
    const int px4 = (rem & 63) * 4;
    const int sy  = hh + row - 1;
    float4 v = make_float4(0.f, 0.f, 0.f, 0.f);
    if ((unsigned)sy < 256u)
      v = *(const float4*)(xb + (size_t)c * HWPROD + (size_t)sy * 256 + px4);
    xs[XS(row, px4 + 1, c)] = (_Float16)v.x;   // +1: halo shift
    xs[XS(row, px4 + 2, c)] = (_Float16)v.y;
    xs[XS(row, px4 + 3, c)] = (_Float16)v.z;
    xs[XS(row, px4 + 4, c)] = (_Float16)v.w;
  }
  if (tid < 3 * 64) { // zero halo columns (SAME padding)
    const int row = tid / 64, c = tid & 63;
    xs[XS(row, 0, c)]   = (_Float16)0.0f;
    xs[XS(row, 257, c)] = (_Float16)0.0f;
  }
  __syncthreads();

  float8 a00 = {}, a01 = {}, a10 = {}, a11 = {}; // acc[mtile][ntile]
  const int px0   = wave * 32;
  const int nlo   = lane & 15;
  const int chalf = (lane < 16) ? 0 : 16;  // B-fragment K-half
  const int rb    = (lane < 16) ? 0 : 8;   // C-fragment M offset

#pragma unroll 1
  for (int tap = 0; tap < 9; ++tap) {
    const int dy = tap / 3, dx = tap - (tap / 3) * 3;
    const int sx0 = px0 + nlo + dx;     // halo-shifted column: sample + 1
    const int sx1 = sx0 + 16;

#pragma unroll
    for (int cb = 0; cb < 2; ++cb) {
      const int ks = tap * 2 + cb;
      const int cbase = cb * 32 + chalf;
      const half16 fa0 = *(const half16*)&lw[LW(0, ks, lane, 0)];
      const half16 fa1 = *(const half16*)&lw[LW(1, ks, lane, 0)];
      const half16 fb0 = *(const half16*)&xs[XS(dy, sx0, cbase)];
      const half16 fb1 = *(const half16*)&xs[XS(dy, sx1, cbase)];
      a00 = wmma_f16(fa0, fb0, a00);
      a01 = wmma_f16(fa0, fb1, a01);
      a10 = wmma_f16(fa1, fb0, a10);
      a11 = wmma_f16(fa1, fb1, a11);
    }

    // heatmap channel (c=64): rank-1 VALU update in C-fragment layout
    const int sy  = hh + dy - 1;
    const int gx0 = px0 + nlo + dx - 1;
    const int gx1 = gx0 + 16;
    const bool rowok = (unsigned)sy < 256u;
    const float mk0 = (rowok && (unsigned)gx0 < 256u) ? 1.0f : 0.0f;
    const float mk1 = (rowok && (unsigned)gx1 < 256u) ? 1.0f : 0.0f;
    const size_t roff = (size_t)(sy & 255) * 256;
    const float hv0 = hb[roff + (gx0 & 255)] * mk0;
    const float hv1 = hb[roff + (gx1 & 255)] * mk1;
#pragma unroll
    for (int r = 0; r < 8; ++r) {
      const float wA = lw64[(r + rb) * 9 + tap];
      const float wB = lw64[(16 + r + rb) * 9 + tap];
      a00[r] += wA * hv0;  a01[r] += wA * hv1;
      a10[r] += wB * hv0;  a11[r] += wB * hv1;
    }
  }

  __syncthreads();  // everyone done reading xs; safe to alias as h1s

  // BN1 + SiLU, transpose C-layout -> pixel-major via LDS
#pragma unroll
  for (int r = 0; r < 8; ++r) {
    const int mA = r + rb, mB = 16 + r + rb;
    h1s[(wave * 32 + nlo) * 32 + mA]      = (_Float16)silu(a00[r] * lsc[mA] + lsh[mA]);
    h1s[(wave * 32 + 16 + nlo) * 32 + mA] = (_Float16)silu(a01[r] * lsc[mA] + lsh[mA]);
    h1s[(wave * 32 + nlo) * 32 + mB]      = (_Float16)silu(a10[r] * lsc[mB] + lsh[mB]);
    h1s[(wave * 32 + 16 + nlo) * 32 + mB] = (_Float16)silu(a11[r] * lsc[mB] + lsh[mB]);
  }
  __syncthreads();

  // 32->2 projection + grid build (one lane per pixel)
  const int p = wave * 32 + lane;
  float o0 = bo2[0], o1 = bo2[1];
#pragma unroll
  for (int c = 0; c < 32; ++c) {
    const float hv = (float)h1s[(wave * 32 + lane) * 32 + c];
    o0 += w2[c]      * hv;
    o1 += w2[32 + c] * hv;
  }
  const float hmv = hb[(size_t)hh * 256 + p];
  const float gx = fminf(1.f, fmaxf(-1.f, (-1.f + (2.f / 255.f) * (float)p)  + o0 * 0.1f * hmv));
  const float gy = fminf(1.f, fmaxf(-1.f, (-1.f + (2.f / 255.f) * (float)hh) + o1 * 0.1f * hmv));
  float2* gp = (float2*)(grid + (((size_t)bb * 256 + hh) * 256 + p) * 2);
  *gp = make_float2(gx, gy);
}

// -------------------------------------------------------------------------
// Kernel 2: grid_sample (fused into B-fragment build) + 1x1(64->64) GEMM
// + BN2 + SiLU + gamma*residual. M = 64 out ch (4 tiles), N = pixels
// (wave = 16 px), K = 64 channels (2 WMMA steps). Gathers are L2-resident
// (x = 134 MB < 192 MB L2); no intermediate ever hits HBM.
// -------------------------------------------------------------------------
__global__ __launch_bounds__(256) void k2_enh(
    const float* __restrict__ x, const float* __restrict__ grid,
    const float* __restrict__ we,
    const float* __restrict__ g2, const float* __restrict__ b2,
    const float* __restrict__ m2, const float* __restrict__ v2,
    const float* __restrict__ gamma, float* __restrict__ out)
{
  __shared__ __align__(32) _Float16 lws[2][4][32][16]; // w_enh A-fragments
  __shared__ float lsc[64], lsh[64];

  const int tid  = threadIdx.x;
  const int lane = tid & 31;
  const int wave = tid >> 5;

  { // one A-fragment entry per thread (2*4*32 == 256)
    const int l  = tid & 31;
    const int mt = (tid >> 5) & 3;
    const int ks = tid >> 7;
    const int m  = mt * 16 + (l & 15);
    const int kh = (l < 16) ? 0 : 16;
#pragma unroll
    for (int j = 0; j < 16; ++j) {
      const int k = j + kh;
      lws[ks][mt][l][j] = (_Float16)we[m * 64 + ks * 32 + k];
    }
  }
  if (tid < 64) {
    const float inv = g2[tid] * rsqrtf(v2[tid] + 1e-5f);
    lsc[tid] = inv;
    lsh[tid] = b2[tid] - m2[tid] * inv;
  }
  __syncthreads();

  const int nlo = lane & 15;
  const size_t P = ((size_t)blockIdx.x * 8 + wave) * 16 + nlo; // global pixel id
  const int bb  = (int)(P >> 16);
  const int pix = (int)(P & 65535);

  // bilinear sampling setup (border padding)
  const float2 g = *(const float2*)(grid + P * 2);
  const float ix = (g.x + 1.f) * 0.5f * 255.f;
  const float iy = (g.y + 1.f) * 0.5f * 255.f;
  const float xf = floorf(ix), yf = floorf(iy);
  const float wx = ix - xf, wy = iy - yf;
  int x0 = (int)xf, y0 = (int)yf;
  int x1 = x0 + 1, y1 = y0 + 1;
  x0 = min(max(x0, 0), 255);  x1 = min(max(x1, 0), 255);
  y0 = min(max(y0, 0), 255);  y1 = min(max(y1, 0), 255);
  const int o00 = y0 * 256 + x0, o01 = y0 * 256 + x1;
  const int o10 = y1 * 256 + x0, o11 = y1 * 256 + x1;
  const float w00 = (1.f - wx) * (1.f - wy), w01 = wx * (1.f - wy);
  const float w10 = (1.f - wx) * wy,         w11 = wx * wy;

  const int chalf = (lane < 16) ? 0 : 16;
  const float* __restrict__ xb = x + (size_t)bb * CC * HWPROD;

  float8 acc0 = {}, acc1 = {}, acc2 = {}, acc3 = {};
#pragma unroll
  for (int ks = 0; ks < 2; ++ks) {
    half16 fb;
#pragma unroll
    for (int j = 0; j < 16; ++j) { // B element j <-> K = chalf+j ; grid-sample inline
      const float* cp = xb + (size_t)(ks * 32 + chalf + j) * HWPROD;
      fb[j] = (_Float16)(cp[o00] * w00 + cp[o01] * w01 + cp[o10] * w10 + cp[o11] * w11);
    }
    acc0 = wmma_f16(*(const half16*)&lws[ks][0][lane][0], fb, acc0);
    acc1 = wmma_f16(*(const half16*)&lws[ks][1][lane][0], fb, acc1);
    acc2 = wmma_f16(*(const half16*)&lws[ks][2][lane][0], fb, acc2);
    acc3 = wmma_f16(*(const half16*)&lws[ks][3][lane][0], fb, acc3);
  }

  // epilogue: BN2 + SiLU + residual, straight from C-fragment layout
  const float gm = gamma[0];
  const int rb = (lane < 16) ? 0 : 8;
#pragma unroll
  for (int r = 0; r < 8; ++r) {
    {
      const int m = r + rb;
      const size_t oi = (size_t)bb * CC * HWPROD + (size_t)m * HWPROD + pix;
      out[oi] = x[oi] + gm * silu(acc0[r] * lsc[m] + lsh[m]);
    }
    {
      const int m = 16 + r + rb;
      const size_t oi = (size_t)bb * CC * HWPROD + (size_t)m * HWPROD + pix;
      out[oi] = x[oi] + gm * silu(acc1[r] * lsc[m] + lsh[m]);
    }
    {
      const int m = 32 + r + rb;
      const size_t oi = (size_t)bb * CC * HWPROD + (size_t)m * HWPROD + pix;
      out[oi] = x[oi] + gm * silu(acc2[r] * lsc[m] + lsh[m]);
    }
    {
      const int m = 48 + r + rb;
      const size_t oi = (size_t)bb * CC * HWPROD + (size_t)m * HWPROD + pix;
      out[oi] = x[oi] + gm * silu(acc3[r] * lsc[m] + lsh[m]);
    }
  }
}

extern "C" void kernel_launch(void* const* d_in, const int* in_sizes, int n_in,
                              void* d_out, int out_size, void* d_ws, size_t ws_size,
                              hipStream_t stream) {
  const float* x    = (const float*)d_in[0];
  const float* hmp  = (const float*)d_in[1];
  const float* w1   = (const float*)d_in[2];
  const float* g1   = (const float*)d_in[3];
  const float* b1   = (const float*)d_in[4];
  const float* m1   = (const float*)d_in[5];
  const float* v1   = (const float*)d_in[6];
  const float* w2   = (const float*)d_in[7];
  const float* bo2  = (const float*)d_in[8];
  const float* we   = (const float*)d_in[9];
  const float* g2   = (const float*)d_in[10];
  const float* b2   = (const float*)d_in[11];
  const float* m2   = (const float*)d_in[12];
  const float* v2   = (const float*)d_in[13];
  const float* gam  = (const float*)d_in[14];
  float* out  = (float*)d_out;
  float* grid = (float*)d_ws; // 8*256*256*2 f32 = 4 MB scratch

  // one block per (batch,row): 8*256 = 2048 blocks, 8 wave32s, ~162 KB LDS
  k1_offset_grid<<<2048, 256, K1_LDS_BYTES, stream>>>(x, hmp, w1, g1, b1, m1, v1, w2, bo2, grid);
  // 524288 pixels / 128 per block = 4096 blocks
  k2_enh<<<4096, 256, 0, stream>>>(x, grid, we, g2, b2, m2, v2, gam, out);
}
